// GAT_3ly_66709432041779
// MI455X (gfx1250) — matrix operand order repaired and checked by reading, not compiled, not run on hardware
//
#include <hip/hip_runtime.h>

typedef __attribute__((ext_vector_type(16))) __bf16 v16bf;
typedef __attribute__((ext_vector_type(8)))  float  v8f;

#define LRELU_SLOPE 0.2f

// =====================================================================
// GEMM: H[N x 64] = X[N x FIN] @ W[FIN x 64]  (bf16 WMMA, f32 accumulate)
// Block = 256 threads = 8 waves; each wave: 16 rows x 64 cols.
// W staged in LDS transposed to B-fragment order: ldsWT[col*FIN + k].
// A tile is software-pipelined: all global loads issued before the
// WMMA loop so s_wait_loadcnt retires them progressively.
// =====================================================================
template<int FIN>
__global__ __launch_bounds__(256) void gemm_wmma(const float* __restrict__ X,
                                                 const float* __restrict__ W,
                                                 float* __restrict__ H, int n)
{
    constexpr int KIT = FIN / 32;
    __shared__ __bf16 ldsWT[64 * FIN];
    const int tid = threadIdx.x;
    for (int i = tid; i < FIN * 64; i += 256) {
        const int k = i >> 6;       // row of W
        const int c = i & 63;       // col of W
        ldsWT[c * FIN + k] = (__bf16)W[i];
    }

    const int lane  = tid & 31;
    const int wave  = tid >> 5;
    const int mbase = blockIdx.x * 128 + wave * 16;
    const int l15   = lane & 15;
    const int hi    = (lane >= 16) ? 1 : 0;

    // A-row this lane reads (clamped so EXEC stays all-ones for WMMA)
    int arow = mbase + l15;
    if (arow >= n) arow = n - 1;
    const float* __restrict__ xrow = X + (size_t)arow * FIN;

    // ---- Issue ALL A-tile loads up front (software pipeline) ----
    float4 axf[KIT][4];
    #pragma unroll
    for (int j = 0; j < KIT; ++j) {
        const int ka = j * 32 + (hi ? 16 : 0);
        axf[j][0] = *(const float4*)&xrow[ka + 0];
        axf[j][1] = *(const float4*)&xrow[ka + 4];
        axf[j][2] = *(const float4*)&xrow[ka + 8];
        axf[j][3] = *(const float4*)&xrow[ka + 12];
    }

    __syncthreads();

    v8f acc0 = {}, acc1 = {}, acc2 = {}, acc3 = {};

    #pragma unroll
    for (int j = 0; j < KIT; ++j) {
        // ---- A fragment (16x32 bf16): convert staged f32 -> bf16
        v16bf a;
        #pragma unroll
        for (int q = 0; q < 4; ++q) {
            const float4 f = axf[j][q];
            a[q * 4 + 0] = (__bf16)f.x;
            a[q * 4 + 1] = (__bf16)f.y;
            a[q * 4 + 2] = (__bf16)f.z;
            a[q * 4 + 3] = (__bf16)f.w;
        }

        // ---- B fragments (32x16 bf16): lane holds one column, 16 consecutive K
        const int kb = j * 32 + (hi ? 16 : 0);
        const v16bf b0 = *(const v16bf*)&ldsWT[( 0 + l15) * FIN + kb];
        const v16bf b1 = *(const v16bf*)&ldsWT[(16 + l15) * FIN + kb];
        const v16bf b2 = *(const v16bf*)&ldsWT[(32 + l15) * FIN + kb];
        const v16bf b3 = *(const v16bf*)&ldsWT[(48 + l15) * FIN + kb];

        acc0 = __builtin_amdgcn_wmma_f32_16x16x32_bf16(false, a, false, b0, (short)0, acc0, false, false);
        acc1 = __builtin_amdgcn_wmma_f32_16x16x32_bf16(false, a, false, b1, (short)0, acc1, false, false);
        acc2 = __builtin_amdgcn_wmma_f32_16x16x32_bf16(false, a, false, b2, (short)0, acc2, false, false);
        acc3 = __builtin_amdgcn_wmma_f32_16x16x32_bf16(false, a, false, b3, (short)0, acc3, false, false);
    }

    // C/D layout: lanes 0-15 N=lane, VGPR r -> M=r ; lanes 16-31 -> M=r+8
    #pragma unroll
    for (int r = 0; r < 8; ++r) {
        const int row = mbase + r + (hi ? 8 : 0);
        if (row < n) {
            float* orow = H + (size_t)row * 64 + l15;
            orow[ 0] = acc0[r];
            orow[16] = acc1[r];
            orow[32] = acc2[r];
            orow[48] = acc3[r];
        }
    }
}

// =====================================================================
// Per-node attention logits: as[n] = h[n].a_src ; ad[n] = h[n].a_dst
// One wave32 per node, 2 channels per lane, shuffle-tree reduce.
// =====================================================================
__global__ __launch_bounds__(256) void alpha_kernel(const float* __restrict__ H,
                                                    const float* __restrict__ a_s,
                                                    const float* __restrict__ a_d,
                                                    float* __restrict__ as_out,
                                                    float* __restrict__ ad_out, int n)
{
    const int node = blockIdx.x * 8 + (threadIdx.x >> 5);
    if (node >= n) return;
    const int lane = threadIdx.x & 31;
    const float* h = H + (size_t)node * 64;
    const float h0 = h[lane], h1 = h[lane + 32];
    float s1 = h0 * a_s[lane] + h1 * a_s[lane + 32];
    float s2 = h0 * a_d[lane] + h1 * a_d[lane + 32];
    #pragma unroll
    for (int off = 16; off > 0; off >>= 1) {
        s1 += __shfl_down(s1, off);
        s2 += __shfl_down(s2, off);
    }
    if (lane == 0) { as_out[node] = s1; ad_out[node] = s2; }
}

// =====================================================================
// Init: m = -inf, s = 0, out = 0
// =====================================================================
__global__ __launch_bounds__(256) void init_kernel(float* __restrict__ m, float* __restrict__ s,
                                                   float* __restrict__ out, int n, int ntot)
{
    const int i = blockIdx.x * blockDim.x + threadIdx.x;
    if (i < n) { m[i] = -__builtin_inff(); s[i] = 0.0f; }
    if (i < ntot) out[i] = 0.0f;
}

__device__ __forceinline__ void edge_ends(const int* __restrict__ ei, int e, int E,
                                          int& src, int& dst)
{
    if (e < E) { src = ei[e]; dst = ei[E + e]; }
    else       { src = e - E; dst = e - E; }      // self-loop
}

__device__ __forceinline__ float edge_val(const float* as, const float* ad, int src, int dst)
{
    const float v = as[src] + ad[dst];
    return (v > 0.0f) ? v : LRELU_SLOPE * v;
}

// Pass 1: segment max over dst (float atomic max via int/uint trick)
__global__ __launch_bounds__(256) void edge_max_kernel(const int* __restrict__ ei,
                                                       const float* __restrict__ as,
                                                       const float* __restrict__ ad,
                                                       float* __restrict__ m, int E, int Etot)
{
    const int e = blockIdx.x * blockDim.x + threadIdx.x;
    if (e >= Etot) return;
    int src, dst; edge_ends(ei, e, E, src, dst);
    const float v = edge_val(as, ad, src, dst);
    if (v >= 0.0f) atomicMax((int*)(m + dst), __float_as_int(v));
    else           atomicMin((unsigned int*)(m + dst), __float_as_uint(v));
}

// Pass 2: segment sum of exp(v - m[dst])
__global__ __launch_bounds__(256) void edge_sum_kernel(const int* __restrict__ ei,
                                                       const float* __restrict__ as,
                                                       const float* __restrict__ ad,
                                                       const float* __restrict__ m,
                                                       float* __restrict__ s, int E, int Etot)
{
    const int e = blockIdx.x * blockDim.x + threadIdx.x;
    if (e >= Etot) return;
    int src, dst; edge_ends(ei, e, E, src, dst);
    const float v = edge_val(as, ad, src, dst);
    atomicAdd(s + dst, __expf(v - m[dst]));
}

// Pass 3: out[dst] += alpha * h[src]
// 16 lanes/edge; group leader computes alpha and broadcasts via shfl,
// each lane moves a float4 of channels (coalesced read + atomics).
__global__ __launch_bounds__(256) void edge_scatter_kernel(const int* __restrict__ ei,
                                                           const float* __restrict__ as,
                                                           const float* __restrict__ ad,
                                                           const float* __restrict__ m,
                                                           const float* __restrict__ s,
                                                           const float* __restrict__ h,
                                                           float* __restrict__ out, int E, int Etot)
{
    const int tid = blockIdx.x * blockDim.x + threadIdx.x;
    const int e   = tid >> 4;
    if (e >= Etot) return;
    const int lane   = threadIdx.x & 31;
    const int c4     = lane & 15;
    const int leader = lane & 16;          // lane 0 or 16 of this wave

    int src = 0, dst = 0; float alpha = 0.0f;
    if (c4 == 0) {
        edge_ends(ei, e, E, src, dst);
        const float v = edge_val(as, ad, src, dst);
        alpha = __expf(v - m[dst]) / (s[dst] + 1e-16f);
    }
    src   = __shfl(src,   leader);
    dst   = __shfl(dst,   leader);
    alpha = __shfl(alpha, leader);

    const float4 hv = ((const float4*)(h + (size_t)src * 64))[c4];
    float* o = out + (size_t)dst * 64 + c4 * 4;
    atomicAdd(o + 0, alpha * hv.x);
    atomicAdd(o + 1, alpha * hv.y);
    atomicAdd(o + 2, alpha * hv.z);
    atomicAdd(o + 3, alpha * hv.w);
}

// out = relu(out + b)  over N*64
__global__ __launch_bounds__(256) void bias_relu_kernel(float* __restrict__ out,
                                                        const float* __restrict__ b, int ntot)
{
    const int i = blockIdx.x * blockDim.x + threadIdx.x;
    if (i >= ntot) return;
    out[i] = fmaxf(out[i] + b[i & 63], 0.0f);
}

// final: d_out[n] = sigmoid(h[n].Wl + bl)  — one wave per node
__global__ __launch_bounds__(256) void final_kernel(const float* __restrict__ H,
                                                    const float* __restrict__ Wl,
                                                    const float* __restrict__ bl,
                                                    float* __restrict__ out, int n)
{
    const int node = blockIdx.x * 8 + (threadIdx.x >> 5);
    if (node >= n) return;
    const int lane = threadIdx.x & 31;
    const float* h = H + (size_t)node * 64;
    float acc = h[lane] * Wl[lane] + h[lane + 32] * Wl[lane + 32];
    #pragma unroll
    for (int off = 16; off > 0; off >>= 1) acc += __shfl_down(acc, off);
    if (lane == 0) {
        const float z = acc + bl[0];
        out[node] = 1.0f / (1.0f + __expf(-z));
    }
}

// =====================================================================
extern "C" void kernel_launch(void* const* d_in, const int* in_sizes, int n_in,
                              void* d_out, int out_size, void* d_ws, size_t ws_size,
                              hipStream_t stream)
{
    const float* x  = (const float*)d_in[0];
    const int*   ei = (const int*)d_in[1];
    const float* W[3]  = { (const float*)d_in[2], (const float*)d_in[6],  (const float*)d_in[10] };
    const float* As[3] = { (const float*)d_in[3], (const float*)d_in[7],  (const float*)d_in[11] };
    const float* Ad[3] = { (const float*)d_in[4], (const float*)d_in[8],  (const float*)d_in[12] };
    const float* Bv[3] = { (const float*)d_in[5], (const float*)d_in[9],  (const float*)d_in[13] };
    const float* Wl = (const float*)d_in[14];
    const float* bl = (const float*)d_in[15];

    const int N    = in_sizes[0] / 128;
    const int E    = in_sizes[1] / 2;
    const int Etot = E + N;
    const int NF   = N * 64;

    float* bufA = (float*)d_ws;                 // h (pre-aggregation features)
    float* bufB = bufA + (size_t)NF;            // layer features / aggregated out
    float* asb  = bufB + (size_t)NF;
    float* adb  = asb + N;
    float* mb   = adb + N;
    float* sb   = mb + N;

    const int TB = 256;
    const dim3 blk(TB);
    const dim3 gGemm((N + 127) / 128);
    const dim3 gNode((N + 7) / 8);
    const dim3 gInit((NF + TB - 1) / TB);
    const dim3 gEdge((Etot + TB - 1) / TB);
    const dim3 gScat(((Etot * 16) + TB - 1) / TB);

    for (int l = 0; l < 3; ++l) {
        const float* in = (l == 0) ? x : bufB;
        if (l == 0) gemm_wmma<128><<<gGemm, blk, 0, stream>>>(in, W[l], bufA, N);
        else        gemm_wmma< 64><<<gGemm, blk, 0, stream>>>(in, W[l], bufA, N);

        alpha_kernel<<<gNode, blk, 0, stream>>>(bufA, As[l], Ad[l], asb, adb, N);
        init_kernel<<<gInit, blk, 0, stream>>>(mb, sb, bufB, N, NF);
        edge_max_kernel<<<gEdge, blk, 0, stream>>>(ei, asb, adb, mb, E, Etot);
        edge_sum_kernel<<<gEdge, blk, 0, stream>>>(ei, asb, adb, mb, sb, E, Etot);
        edge_scatter_kernel<<<gScat, blk, 0, stream>>>(ei, asb, adb, mb, sb, bufA, bufB, E, Etot);
        bias_relu_kernel<<<gInit, blk, 0, stream>>>(bufB, Bv[l], NF);
    }

    final_kernel<<<gNode, blk, 0, stream>>>(bufB, Wl, bl, (float*)d_out, N);
}
